// PairUpdate_15564961481076
// MI455X (gfx1250) — matrix operand order repaired and checked by reading, not compiled, not run on hardware
//
#include <hip/hip_runtime.h>

typedef _Float16 f16;
typedef _Float16 v16h __attribute__((ext_vector_type(16)));
typedef _Float16 h2   __attribute__((ext_vector_type(2)));
typedef float    v8f  __attribute__((ext_vector_type(8)));

constexpr int DN = 256;   // D_NODE
constexpr int DP = 128;   // D_PAIR
constexpr int LL = 512;   // L
constexpr int BB = 2;     // B
constexpr float LN_EPS = 1e-5f;

__device__ __forceinline__ unsigned pkmul_f16x2(unsigned a, unsigned b) {
  union { unsigned u; h2 h; } x, y, r;
  x.u = a; y.u = b;
  r.h = x.h * y.h;          // -> v_pk_mul_f16
  return r.u;
}

// ---------------------------------------------------------------------------
// Kernel 1: layernorm(single) then a = x@Wa.T+ba, b = x@Wb.T+bb, stored f16.
// One block per (b,l) row. 256 threads.
// ---------------------------------------------------------------------------
__global__ __launch_bounds__(256) void k_ln_proj(
    const float* __restrict__ single, const float* __restrict__ ng, const float* __restrict__ nb,
    const float* __restrict__ Wa, const float* __restrict__ ba,
    const float* __restrict__ Wb, const float* __restrict__ bb,
    f16* __restrict__ a_out, f16* __restrict__ b_out)
{
  __shared__ float red[256];
  __shared__ float sx[256];
  const int t = threadIdx.x;
  const int row = blockIdx.x;                    // 0 .. B*L-1
  float x = single[(size_t)row * DN + t];

  red[t] = x; __syncthreads();
  for (int s = 128; s > 0; s >>= 1) { if (t < s) red[t] += red[t + s]; __syncthreads(); }
  float mu = red[0] * (1.f / 256.f);
  __syncthreads();
  float d = x - mu;
  red[t] = d * d; __syncthreads();
  for (int s = 128; s > 0; s >>= 1) { if (t < s) red[t] += red[t + s]; __syncthreads(); }
  float rstd = rsqrtf(red[0] * (1.f / 256.f) + LN_EPS);
  sx[t] = d * rstd * ng[t] + nb[t];
  __syncthreads();

  const float* W    = (t < 128) ? Wa : Wb;
  const float* bias = (t < 128) ? ba : bb;
  const int p = t & 127;
  const float* wrow = W + (size_t)p * DN;
  float acc = bias[p];
  #pragma unroll 8
  for (int k = 0; k < DN; ++k) acc += sx[k] * wrow[k];
  f16* dst = (t < 128) ? a_out : b_out;
  dst[(size_t)row * DP + p] = (f16)acc;
}

// ---------------------------------------------------------------------------
// Kernel 2: f32 -> f16 conversion (for Wo)
// ---------------------------------------------------------------------------
__global__ __launch_bounds__(256) void k_cvt(const float* __restrict__ src,
                                             f16* __restrict__ dst, int n)
{
  int i = blockIdx.x * 256 + threadIdx.x;
  if (i < n) dst[i] = (f16)src[i];
}

// ---------------------------------------------------------------------------
// Kernel 3: fused outer-product + Wo GEMM (WMMA f16) + residual + layernorm.
// Block = 16i x 16j tile, 256 threads = 8 waves; wave w handles j = 2w, 2w+1.
// ---------------------------------------------------------------------------
__global__ __launch_bounds__(256) void k_pair(
    const float* __restrict__ pair, const f16* __restrict__ a16, const f16* __restrict__ b16,
    const f16* __restrict__ wo16, const float* __restrict__ bo,
    const float* __restrict__ pg, const float* __restrict__ pb,
    float* __restrict__ out)
{
  extern __shared__ unsigned char smem[];
  // padded row strides (u32): Wo/a/b rows = 64 data + 4 pad = 68; upd rows = 128 + 4 = 132
  unsigned* s_wo = (unsigned*)smem;          // 128 * 68 u32  (Wo, f16 pairs)
  unsigned* s_a  = s_wo + 128 * 68;          // 16  * 68 u32
  unsigned* s_b  = s_a  + 16 * 68;           // 16  * 68 u32
  float*    s_upd = (float*)(s_b + 16 * 68); // 256 * 132 f32

  const int t    = threadIdx.x;
  const int lane = t & 31;
  const int wave = t >> 5;
  const int i0 = blockIdx.x * 16;
  const int j0 = blockIdx.y * 16;
  const int bz = blockIdx.z;

  // ---- stage Wo / a-tile / b-tile into LDS ----
  {
    const unsigned* g = (const unsigned*)wo16;                       // 8192 u32
    #pragma unroll
    for (int it = 0; it < 32; ++it) {
      int idx = t + 256 * it;
      s_wo[(idx >> 6) * 68 + (idx & 63)] = g[idx];
    }
    const unsigned* gaa = (const unsigned*)(a16 + ((size_t)bz * LL + i0) * DP); // 1024 u32
    const unsigned* gbb = (const unsigned*)(b16 + ((size_t)bz * LL + j0) * DP);
    #pragma unroll
    for (int it = 0; it < 4; ++it) {
      int idx = t + 256 * it;
      s_a[(idx >> 6) * 68 + (idx & 63)] = gaa[idx];
      s_b[(idx >> 6) * 68 + (idx & 63)] = gbb[idx];
    }
  }
  float bo_v[8];
  #pragma unroll
  for (int pc = 0; pc < 8; ++pc) bo_v[pc] = bo[pc * 16 + (lane & 15)];
  __syncthreads();

  union V16 { v16h v; unsigned u[8]; uint4 q[2]; };

  #pragma unroll
  for (int jj = 0; jj < 2; ++jj) {
    const int jl = wave * 2 + jj;

    // ---- A operand: a[i,:] * b[j,:] in 16x32 f16 WMMA A-layout, 4 K-windows ----
    V16 A[4];
    #pragma unroll
    for (int kw = 0; kw < 4; ++kw) {
      const int offa = (lane & 15) * 68 + kw * 16 + (lane >> 4) * 4;
      const int offb = jl * 68        + kw * 16 + (lane >> 4) * 4;
      uint4 qa0 = *(const uint4*)(s_a + offa);
      uint4 qa1 = *(const uint4*)(s_a + offa + 8);
      uint4 qb0 = *(const uint4*)(s_b + offb);
      uint4 qb1 = *(const uint4*)(s_b + offb + 8);
      A[kw].u[0] = pkmul_f16x2(qa0.x, qb0.x); A[kw].u[1] = pkmul_f16x2(qa0.y, qb0.y);
      A[kw].u[2] = pkmul_f16x2(qa0.z, qb0.z); A[kw].u[3] = pkmul_f16x2(qa0.w, qb0.w);
      A[kw].u[4] = pkmul_f16x2(qa1.x, qb1.x); A[kw].u[5] = pkmul_f16x2(qa1.y, qb1.y);
      A[kw].u[6] = pkmul_f16x2(qa1.z, qb1.z); A[kw].u[7] = pkmul_f16x2(qa1.w, qb1.w);
    }

    v8f acc[8];
    #pragma unroll
    for (int pc = 0; pc < 8; ++pc) {
      #pragma unroll
      for (int r = 0; r < 8; ++r) acc[pc][r] = bo_v[pc];   // fold in bo
    }

    // ---- D += A(16x32) x B(32x16 = Wo^T chunk) over 4 K-windows, 8 p-chunks ----
    #pragma unroll
    for (int kw = 0; kw < 4; ++kw) {
      #pragma unroll
      for (int pc = 0; pc < 8; ++pc) {
        V16 Bop;
        const int offw = (pc * 16 + (lane & 15)) * 68 + kw * 16 + (lane >> 4) * 4;
        Bop.q[0] = *(const uint4*)(s_wo + offw);
        Bop.q[1] = *(const uint4*)(s_wo + offw + 8);
        acc[pc] = __builtin_amdgcn_wmma_f32_16x16x32_f16(
            false, A[kw].v, false, Bop.v, (short)0, acc[pc], false, false);
      }
    }

    // ---- scatter D (i x p) for this j into s_upd ----
    #pragma unroll
    for (int pc = 0; pc < 8; ++pc) {
      const int p = pc * 16 + (lane & 15);
      #pragma unroll
      for (int r = 0; r < 8; ++r) {
        const int irow = r + 8 * (lane >> 4);
        s_upd[(irow * 16 + jl) * 132 + p] = acc[pc][r];
      }
    }
  }
  __syncthreads();

  // ---- phase 2: one thread per (i,j): residual add + layernorm over 128 p ----
  {
    const int i = t >> 4, j = t & 15;
    const size_t gbase = (((size_t)bz * LL + (i0 + i)) * LL + (j0 + j)) * DP;
    const float4* pr = (const float4*)(pair + gbase);
    float4* po = (float4*)(out + gbase);
    float* srow = s_upd + t * 132;

    float sum = 0.f, sq = 0.f;
    #pragma unroll 4
    for (int q = 0; q < 32; ++q) {
      float4 pv = pr[q];
      float4 uv = *(float4*)(srow + q * 4);
      float v0 = pv.x + uv.x, v1 = pv.y + uv.y, v2 = pv.z + uv.z, v3 = pv.w + uv.w;
      sum += (v0 + v1) + (v2 + v3);
      sq  += (v0 * v0 + v1 * v1) + (v2 * v2 + v3 * v3);
      *(float4*)(srow + q * 4) = make_float4(v0, v1, v2, v3);
    }
    float mu   = sum * (1.f / 128.f);
    float var  = sq * (1.f / 128.f) - mu * mu;
    float rstd = rsqrtf(var + LN_EPS);
    #pragma unroll 4
    for (int q = 0; q < 32; ++q) {
      float4 uv = *(float4*)(srow + q * 4);
      float4 g4 = *(const float4*)(pg + q * 4);
      float4 b4 = *(const float4*)(pb + q * 4);
      po[q] = make_float4((uv.x - mu) * rstd * g4.x + b4.x,
                          (uv.y - mu) * rstd * g4.y + b4.y,
                          (uv.z - mu) * rstd * g4.z + b4.z,
                          (uv.w - mu) * rstd * g4.w + b4.w);
    }
  }
}

// ---------------------------------------------------------------------------
extern "C" void kernel_launch(void* const* d_in, const int* in_sizes, int n_in,
                              void* d_out, int out_size, void* d_ws, size_t ws_size,
                              hipStream_t stream) {
  const float* single = (const float*)d_in[0];
  const float* pair   = (const float*)d_in[1];
  const float* ng     = (const float*)d_in[2];
  const float* nb     = (const float*)d_in[3];
  const float* Wa     = (const float*)d_in[4];
  const float* ba     = (const float*)d_in[5];
  const float* Wb     = (const float*)d_in[6];
  const float* bb     = (const float*)d_in[7];
  const float* Wo     = (const float*)d_in[8];
  const float* bo     = (const float*)d_in[9];
  const float* pg     = (const float*)d_in[10];
  const float* pb     = (const float*)d_in[11];
  float* out = (float*)d_out;

  f16* a16  = (f16*)d_ws;                          // B*L*DP f16
  f16* b16  = a16 + (size_t)BB * LL * DP;
  f16* wo16 = b16 + (size_t)BB * LL * DP;          // DP*DP f16

  k_ln_proj<<<BB * LL, 256, 0, stream>>>(single, ng, nb, Wa, ba, Wb, bb, a16, b16);
  k_cvt<<<(DP * DP + 255) / 256, 256, 0, stream>>>(Wo, wo16, DP * DP);

  const size_t lds_bytes = (size_t)(128 * 68 + 16 * 68 + 16 * 68) * 4  // Wo + a + b
                         + (size_t)256 * 132 * 4;                      // upd staging
  k_pair<<<dim3(LL / 16, LL / 16, BB), 256, lds_bytes, stream>>>(
      pair, a16, b16, wo16, bo, pg, pb, out);
}
